// Attention_52733608460767
// MI455X (gfx1250) — matrix operand order repaired and checked by reading, not compiled, not run on hardware
//
#include <hip/hip_runtime.h>
#include <hip/hip_bf16.h>

// ---- problem constants (from reference) ----
#define B_   4
#define N_   1024
#define E_   1024
#define H_   16
#define D_   64
#define M_   (B_ * N_)          // 4096 rows in the projection GEMMs
#define BH_  (B_ * H_)          // 64 (b,h) pairs
#define INV_SCALE 0.03125f      // 1/sqrt(E) = 1/32

typedef __bf16 bf16;
typedef bf16  v16bf __attribute__((ext_vector_type(16)));
typedef bf16  v8bf  __attribute__((ext_vector_type(8)));
typedef float v8f   __attribute__((ext_vector_type(8)));

// =====================================================================
// Kernel 1: projection  C = X @ W^T + bias, output split into heads.
//   X: [4096 x 1024] f32 (row-major), W: [1024 x 1024] f32 (row-major)
//   C[m,e] = sum_k X[m,k] * W[e,k]   (both K-contiguous: ideal for WMMA)
// Block: 256 threads (8 waves). Block tile: 16 rows x 128 cols.
// Each wave owns one 16x16 tile; A tile shared via LDS as bf16.
// Output layout (b,h,n,d):  b=m>>10, n=m&1023, h=e>>6, d=e&63.
// =====================================================================
__global__ __launch_bounds__(256) void proj_kernel(
    const float* __restrict__ X, const float* __restrict__ W,
    const float* __restrict__ bias,
    bf16* __restrict__ obf, float* __restrict__ of32)
{
    __shared__ __align__(16) bf16 Atile[16][32];

    const int lane    = threadIdx.x & 31;
    const int wave    = threadIdx.x >> 5;
    const int rowBase = blockIdx.x * 16;                // 0..4080
    const int colBase = blockIdx.y * 128 + wave * 16;   // 0..1008

    // A-fragment lane mapping (16-bit A 16x32): row m, K contiguous like B
    const int mA  = lane & 15;
    const int kbA = (lane < 16) ? 0 : 16;
    // B-fragment lane mapping (16-bit B 32x16): lane n, K contiguous
    const int nB  = colBase + (lane & 15);
    const int kbB = (lane < 16) ? 0 : 16;

    v8f acc = {};
    for (int k0 = 0; k0 < E_; k0 += 32) {
        __syncthreads();    // protect previous iteration's reads
        {   // cooperative A tile load: 16x32 f32 -> bf16, 2 elems/thread
            const int idx = threadIdx.x * 2;
            const int r = idx >> 5, c = idx & 31;
            const float* src = X + (size_t)(rowBase + r) * E_ + k0 + c;
            Atile[r][c]     = (bf16)src[0];
            Atile[r][c + 1] = (bf16)src[1];
        }
        __syncthreads();

        // A fragment from LDS (two aligned 16B loads)
        v8bf alo = *(const v8bf*)&Atile[mA][kbA];
        v8bf ahi = *(const v8bf*)&Atile[mA][kbA + 8];
        v16bf a;
        #pragma unroll
        for (int i = 0; i < 8; ++i) { a[i] = alo[i]; a[i + 8] = ahi[i]; }

        // B fragment: W row nB, 16 consecutive K values, cvt f32->bf16
        const float* wsrc = W + (size_t)nB * E_ + k0 + kbB;
        v16bf b;
        #pragma unroll
        for (int i = 0; i < 16; ++i) b[i] = (bf16)wsrc[i];

        acc = __builtin_amdgcn_wmma_f32_16x16x32_bf16(
                  false, a, false, b, (short)0, acc, false, false);
    }

    // bias add + store (D layout: lane=N, vgpr r -> M=r or r+8)
    const int eg = colBase + (lane & 15);
    const float bv = bias[eg];
    const int h = eg >> 6, d = eg & 63;
    #pragma unroll
    for (int r = 0; r < 8; ++r) {
        const int mLocal = (lane < 16) ? r : r + 8;
        const int m = rowBase + mLocal;
        const int b = m >> 10, n = m & 1023;
        const size_t oidx = ((size_t)(b * H_ + h) * N_ + n) * D_ + d;
        const float val = acc[r] + bv;
        if (obf)  obf[oidx]  = (bf16)val;
        if (of32) of32[oidx] = val;
    }
}

// =====================================================================
// Kernel 2: fused scores + softmax + (rowsum * Vp) for one (b,h,16-row)
// stripe. 8 waves; wave w computes 16x16 score tiles at cols w*16+j*128
// (K=64 -> two bf16 WMMAs each). Scores scaled and parked in LDS
// (16 x 1024, ~65KB: legal, CDNA5 allows 320KB LDS per workgroup).
// Softmax: 16 lanes per row, width-16 __shfl_xor reductions.
// =====================================================================
__global__ __launch_bounds__(256) void attn_kernel(
    const bf16* __restrict__ Qp, const bf16* __restrict__ Kp,
    float* __restrict__ attnOut, float* __restrict__ outV)
{
    __shared__ __align__(16) float Sm[16][N_ + 16];   // pad: stagger rows

    const int lane    = threadIdx.x & 31;
    const int wave    = threadIdx.x >> 5;
    const int bh      = blockIdx.y;          // 0..63
    const int rowBase = blockIdx.x * 16;     // 0..1008
    const size_t qkBase = (size_t)bh * N_ * D_;

    // ---- A fragments: Q rows (reused across all 8 column tiles) ----
    const int mA  = lane & 15;
    const int kbA = (lane < 16) ? 0 : 16;
    const bf16* qrow = Qp + qkBase + (size_t)(rowBase + mA) * D_;
    v16bf a0, a1;
    {
        v8bf lo = *(const v8bf*)(qrow + kbA);
        v8bf hi = *(const v8bf*)(qrow + kbA + 8);
        #pragma unroll
        for (int i = 0; i < 8; ++i) { a0[i] = lo[i]; a0[i + 8] = hi[i]; }
        lo = *(const v8bf*)(qrow + 32 + kbA);
        hi = *(const v8bf*)(qrow + 32 + kbA + 8);
        #pragma unroll
        for (int i = 0; i < 8; ++i) { a1[i] = lo[i]; a1[i + 8] = hi[i]; }
    }

    // ---- score tiles -> LDS ----
    const int nLoc = lane & 15;
    const int kbB  = (lane < 16) ? 0 : 16;
    #pragma unroll
    for (int t = 0; t < 8; ++t) {
        const int col = wave * 16 + t * 128;
        const bf16* krow = Kp + qkBase + (size_t)(col + nLoc) * D_;
        v16bf b0 = *(const v16bf*)(krow + kbB);        // K = 0..31 slice
        v16bf b1 = *(const v16bf*)(krow + 32 + kbB);   // K = 32..63 slice
        v8f s = {};
        s = __builtin_amdgcn_wmma_f32_16x16x32_bf16(
                false, a0, false, b0, (short)0, s, false, false);
        s = __builtin_amdgcn_wmma_f32_16x16x32_bf16(
                false, a1, false, b1, (short)0, s, false, false);
        #pragma unroll
        for (int r = 0; r < 8; ++r) {
            const int mLocal = (lane < 16) ? r : r + 8;
            Sm[mLocal][col + nLoc] = s[r] * INV_SCALE;
        }
    }
    __syncthreads();

    // ---- softmax: thread t -> row = t>>4, 64-col segment = t&15 ----
    const int row  = threadIdx.x >> 4;
    const int seg  = threadIdx.x & 15;
    const int rowG = rowBase + row;
    const float4* rseg = (const float4*)&Sm[row][seg * 64];

    float mx = -3.402823466e38f;
    #pragma unroll
    for (int i = 0; i < 16; ++i) {
        const float4 v = rseg[i];
        mx = fmaxf(mx, fmaxf(fmaxf(v.x, v.y), fmaxf(v.z, v.w)));
    }
    #pragma unroll
    for (int m = 8; m >= 1; m >>= 1) mx = fmaxf(mx, __shfl_xor(mx, m, 16));

    float sum = 0.f;
    #pragma unroll
    for (int i = 0; i < 16; ++i) {
        const float4 v = rseg[i];
        sum += __expf(v.x - mx) + __expf(v.y - mx)
             + __expf(v.z - mx) + __expf(v.w - mx);
    }
    #pragma unroll
    for (int m = 8; m >= 1; m >>= 1) sum += __shfl_xor(sum, m, 16);
    const float inv = 1.0f / sum;

    // normalize, write attn (268MB stream: the bandwidth floor), track rowsum
    float4* dst = (float4*)(attnOut + ((size_t)bh * N_ + rowG) * N_ + seg * 64);
    float ws = 0.f;
    #pragma unroll
    for (int i = 0; i < 16; ++i) {
        const float4 v = rseg[i];
        float4 p;
        p.x = __expf(v.x - mx) * inv;  p.y = __expf(v.y - mx) * inv;
        p.z = __expf(v.z - mx) * inv;  p.w = __expf(v.w - mx) * inv;
        ws += p.x + p.y + p.z + p.w;
        dst[i] = p;
    }
    #pragma unroll
    for (int m = 8; m >= 1; m >>= 1) ws += __shfl_xor(ws, m, 16);

    // out = rowsum(attn) * Vp  (Vp already in out region from proj pass)
    float* ov = outV + (size_t)bh * N_ * D_ + (size_t)rowG * D_ + seg * 4;
    #pragma unroll
    for (int j = 0; j < 4; ++j) ov[j] *= ws;
}

// =====================================================================
// Host launcher. d_in order: q,k,v,Wq,bq,Wk,bk,Wv,bv (all f32).
// d_out: out (B,H,N,D) f32 then attn (B,H,N,N) f32.
// d_ws: Qp/Kp bf16 projections, 16 MB total.
// =====================================================================
extern "C" void kernel_launch(void* const* d_in, const int* in_sizes, int n_in,
                              void* d_out, int out_size, void* d_ws, size_t ws_size,
                              hipStream_t stream) {
    const float* q  = (const float*)d_in[0];
    const float* k  = (const float*)d_in[1];
    const float* v  = (const float*)d_in[2];
    const float* Wq = (const float*)d_in[3];
    const float* bq = (const float*)d_in[4];
    const float* Wk = (const float*)d_in[5];
    const float* bk = (const float*)d_in[6];
    const float* Wv = (const float*)d_in[7];
    const float* bv = (const float*)d_in[8];

    bf16* Qp = (bf16*)d_ws;
    bf16* Kp = Qp + (size_t)BH_ * N_ * D_;            // +4M bf16 (8 MB)

    float* outV = (float*)d_out;                      // (B,H,N,D)
    float* attn = outV + (size_t)BH_ * N_ * D_;       // (B,H,N,N)

    const dim3 pgrid(M_ / 16, E_ / 128);              // 256 x 8
    proj_kernel<<<pgrid, 256, 0, stream>>>(q, Wq, bq, Qp, nullptr);
    proj_kernel<<<pgrid, 256, 0, stream>>>(k, Wk, bk, Kp, nullptr);
    proj_kernel<<<pgrid, 256, 0, stream>>>(v, Wv, bv, nullptr, outV);

    const dim3 agrid(N_ / 16, BH_);                   // 64 x 64
    attn_kernel<<<agrid, 256, 0, stream>>>(Qp, Kp, attn, outV);
}
